// CustomHead_58445914964325
// MI455X (gfx1250) — matrix operand order repaired and checked by reading, not compiled, not run on hardware
//
#include <hip/hip_runtime.h>
#include <hip/hip_bf16.h>

#define DI __device__ __forceinline__

typedef __attribute__((ext_vector_type(16))) __bf16 v16bf;
typedef __attribute__((ext_vector_type(8)))  float  v8f;
typedef int vi4 __attribute__((vector_size(16)));   // int4 for async-LDS builtin

union Frag {
  v16bf v;
  unsigned int u[8];
};

static constexpr int Bc = 32;     // batch
static constexpr int T  = 2048;   // sequence length
static constexpr int C  = 1024;   // n_embd
static constexpr int H  = 128;    // head size
static constexpr int BT = Bc * T; // 65536 rows

// async global->LDS copy path (gfx1250), with fallback to manual staging
#if defined(__has_builtin)
#if __has_builtin(__builtin_amdgcn_global_load_async_to_lds_b128) && \
    __has_builtin(__builtin_amdgcn_s_wait_asynccnt)
#define HAVE_ASYNC_LDS 1
#endif
#endif
#ifndef HAVE_ASYNC_LDS
#define HAVE_ASYNC_LDS 0
#endif

// f32 -> bf16 via hardware conversion
DI unsigned short f2b(float f) {
  __bf16 b = (__bf16)f;
  return __builtin_bit_cast(unsigned short, b);
}

// pack two f32 -> 2x bf16 in one dword (prefer packed hw instruction)
DI unsigned int pack_bf16(float lo, float hi) {
#if defined(__has_builtin) && __has_builtin(__builtin_amdgcn_cvt_pk_bf16_f32)
  auto r = __builtin_amdgcn_cvt_pk_bf16_f32(lo, hi);
  return __builtin_bit_cast(unsigned int, r);
#else
  return (unsigned int)f2b(lo) | ((unsigned int)f2b(hi) << 16);
#endif
}

// ---------------------------------------------------------------------------
// Pass 0: convert W matrices (3 x [H, C] f32) to bf16 once.
// wbf layout: [3][H][C] bf16, stored as uint pairs.
// ---------------------------------------------------------------------------
__global__ __launch_bounds__(256) void cvt_w_kernel(
    const float* __restrict__ Wk, const float* __restrict__ Wq,
    const float* __restrict__ Wv, unsigned int* __restrict__ wbf)
{
  int i = blockIdx.x * 256 + threadIdx.x;     // pair index, 0 .. 3*H*C/2 - 1
  constexpr int PAIRS = H * C / 2;            // per matrix
  int mat = i / PAIRS;
  int off = i - mat * PAIRS;
  const float* W = (mat == 0) ? Wk : (mat == 1 ? Wq : Wv);
  float2 f = *(const float2*)(W + (size_t)off * 2);
  wbf[i] = pack_bf16(f.x, f.y);
}

// ---------------------------------------------------------------------------
// Pass 1: QKV projection.  out[wsel][m, n] = sum_k x[m,k] * W[n,k]  (bf16 out)
// x read by both waves of an M-tile: each wave holds 12 of the 24 N-tiles (K:8 Q:8 V:8).
// A-frag (16x32 bf16): lane half sel, VGPR v -> K = sel*16+2v, M = lane&15
// B-frag (32x16 bf16): lane half sel, VGPR v -> K = sel*16+2v, N = lane&15
// C/D   (16x16 f32)  : lane half sel, VGPR v -> M = sel*8+v,   N = lane&15
// B-fragments are straight dword loads from pre-converted bf16 weights.
// ---------------------------------------------------------------------------
__global__ __launch_bounds__(256) void qkv_proj_kernel(
    const float* __restrict__ x, const unsigned int* __restrict__ wbf,
    unsigned short* __restrict__ qkv)
{
  const int wave  = threadIdx.x >> 5;
  const int lane  = threadIdx.x & 31;
  const int lh    = lane & 15;
  const int sel   = lane >> 4;
  const int m0    = (blockIdx.x * 4 + (wave >> 1)) * 16;
  const int waveN = wave & 1;

  const float* xrow = x + (size_t)(m0 + lh) * C;

  // bf16 weight row pointers (uint = 2 consecutive k elements)
  const unsigned int* wrow[12];
  #pragma unroll
  for (int i = 0; i < 12; ++i) {
    int ntg  = waveN * 12 + i;
    int wsel = ntg >> 3;
    int n    = (ntg & 7) * 16 + lh;
    wrow[i] = wbf + ((size_t)(wsel * H + n) * C) / 2;
  }

  v8f zero = {};
  v8f acc[12];
  #pragma unroll
  for (int i = 0; i < 12; ++i) acc[i] = zero;

  for (int kk = 0; kk < C; kk += 32) {
    const int kk2 = kk / 2;     // uint index
    Frag a;
    #pragma unroll
    for (int v = 0; v < 8; ++v) {
      int ka = sel * 16 + v * 2;
      float2 f = *(const float2*)(xrow + kk + ka);
      a.u[v] = pack_bf16(f.x, f.y);
    }
    #pragma unroll
    for (int i = 0; i < 12; ++i) {
      Frag b;
      #pragma unroll
      for (int v = 0; v < 8; ++v)
        b.u[v] = wrow[i][kk2 + sel * 8 + v];   // K = sel*16 + 2v (pair)
      acc[i] = __builtin_amdgcn_wmma_f32_16x16x32_bf16(
          false, a.v, false, b.v, (short)0, acc[i], false, false);
    }
  }

  #pragma unroll
  for (int i = 0; i < 12; ++i) {
    int ntg  = waveN * 12 + i;
    int wsel = ntg >> 3;
    int n0   = (ntg & 7) * 16;
    unsigned short* o = qkv + ((size_t)wsel * BT + m0) * H + n0;
    #pragma unroll
    for (int v = 0; v < 8; ++v)
      o[(size_t)(sel * 8 + v) * H + lh] = f2b(acc[i][v]);
  }
}

// ---------------------------------------------------------------------------
// Pass 2: causal flash attention over bf16 Q,K,V.
// Block = 4 waves = 64 query rows; each wave owns one 16-row tile, Q in regs.
// Per 32-key block: stage K via async global->LDS DMA (row-major) and V
// (transposed, manual), then S = Q K^T (8 wmma), online softmax (shfl
// reductions over 16-lane groups), P via LDS round-trip, O += P V (8 wmma).
// ---------------------------------------------------------------------------
__global__ __launch_bounds__(128) void flash_attn_kernel(
    const unsigned short* __restrict__ qkv, float* __restrict__ out)
{
  __shared__ unsigned short Ktile[32 * 128];   // [key][h]
  __shared__ unsigned short Vt[128 * 32];      // [h][key] (transposed)
  __shared__ unsigned short Pst[4][16 * 32];   // per-wave P staging

  const int tid  = threadIdx.x;
  const int wave = tid >> 5;
  const int lane = tid & 31;
  const int lh   = lane & 15;
  const int sel  = lane >> 4;
  const int b    = blockIdx.y;
  const int q0   = blockIdx.x * 64 + wave * 16;

  const unsigned short* Kb = qkv + (size_t)b * T * H;
  const unsigned short* Qb = qkv + (size_t)BT * H + (size_t)b * T * H;
  const unsigned short* Vb = qkv + 2 * (size_t)BT * H + (size_t)b * T * H;

  // Q A-fragments: rows q0..q0+15, 4 chunks of 32 along h
  Frag aq[4];
  {
    const unsigned short* qrow = Qb + (size_t)(q0 + lh) * H;
    #pragma unroll
    for (int c = 0; c < 4; ++c)
      #pragma unroll
      for (int v = 0; v < 8; ++v) {
        int ka = c * 32 + sel * 16 + v * 2;
        aq[c].u[v] = *(const unsigned int*)(qrow + ka);
      }
  }

  v8f zero = {};
  v8f o[8];
  #pragma unroll
  for (int nt = 0; nt < 8; ++nt) o[nt] = zero;
  float mprev[8], lsum[8];
  #pragma unroll
  for (int v = 0; v < 8; ++v) { mprev[v] = -1e30f; lsum[v] = 0.f; }

  const int   kb_count = blockIdx.x * 2 + 2;
  const float scale    = 0.03125f;            // C^-0.5 (reference scaling)

  for (int kb = 0; kb < kb_count; ++kb) {
    const int j0 = kb * 32;
    __syncthreads();

    // stage K rows [j0, j0+32) x 128h into Ktile
#if HAVE_ASYNC_LDS
    #pragma unroll
    for (int i = 0; i < 4; ++i) {
      int idx = tid + 128 * i;
      int row = idx >> 4;
      int qq  = idx & 15;
      const unsigned short* g = Kb + (size_t)(j0 + row) * H + qq * 8;
      unsigned short*       l = &Ktile[row * 128 + qq * 8];
      __builtin_amdgcn_global_load_async_to_lds_b128(
          (__attribute__((address_space(1))) vi4*)g,
          (__attribute__((address_space(3))) vi4*)l, 0, 0);
    }
#else
    #pragma unroll
    for (int i = 0; i < 4; ++i) {
      int idx = tid + 128 * i;
      int row = idx >> 4;
      int qq  = idx & 15;
      uint4 d = *(const uint4*)(Kb + (size_t)(j0 + row) * H + qq * 8);
      *(uint4*)&Ktile[row * 128 + qq * 8] = d;
    }
#endif
    // stage V transposed: Vt[h][key]  (async DMA cannot transpose)
    #pragma unroll
    for (int i = 0; i < 4; ++i) {
      int idx = tid + 128 * i;
      int row = idx >> 4;
      int qq  = idx & 15;
      union { uint4 q; unsigned short s[8]; } d;
      d.q = *(const uint4*)(Vb + (size_t)(j0 + row) * H + qq * 8);
      #pragma unroll
      for (int j = 0; j < 8; ++j)
        Vt[(qq * 8 + j) * 32 + row] = d.s[j];
    }
#if HAVE_ASYNC_LDS
    __builtin_amdgcn_s_wait_asynccnt(0);
#endif
    __syncthreads();

    if (j0 > q0 + 15) continue;   // fully masked for this wave (barriers stay uniform)

    // S = Q K^T : two 16-key tiles, accumulate over 4 h-chunks
    v8f s0 = zero, s1 = zero;
    #pragma unroll
    for (int c = 0; c < 4; ++c) {
      Frag b0, b1;
      #pragma unroll
      for (int v = 0; v < 8; ++v) {
        int kb2 = c * 32 + sel * 16 + v * 2;
        b0.u[v] = *(const unsigned int*)&Ktile[(0  + lh) * 128 + kb2];
        b1.u[v] = *(const unsigned int*)&Ktile[(16 + lh) * 128 + kb2];
      }
      s0 = __builtin_amdgcn_wmma_f32_16x16x32_bf16(false, aq[c].v, false, b0.v, (short)0, s0, false, false);
      s1 = __builtin_amdgcn_wmma_f32_16x16x32_bf16(false, aq[c].v, false, b1.v, (short)0, s1, false, false);
    }

    // causal mask + online softmax; row m = q0 + sel*8 + v lives in one
    // 16-lane half, so shfl_xor(...,16-wide) reductions match the C/D layout.
    float p0[8], p1[8];
    #pragma unroll
    for (int v = 0; v < 8; ++v) {
      int mg = q0 + sel * 8 + v;
      float a0 = (j0 + lh      <= mg) ? s0[v] * scale : -1e30f;
      float a1 = (j0 + 16 + lh <= mg) ? s1[v] * scale : -1e30f;
      float mv = fmaxf(a0, a1);
      #pragma unroll
      for (int off = 8; off > 0; off >>= 1)
        mv = fmaxf(mv, __shfl_xor(mv, off, 16));
      float mnew  = fmaxf(mprev[v], mv);
      float alpha = __expf(mprev[v] - mnew);
      float e0 = __expf(a0 - mnew);
      float e1 = __expf(a1 - mnew);
      float r  = e0 + e1;
      #pragma unroll
      for (int off = 8; off > 0; off >>= 1)
        r += __shfl_xor(r, off, 16);
      lsum[v]  = lsum[v] * alpha + r;
      mprev[v] = mnew;
      #pragma unroll
      for (int nt = 0; nt < 8; ++nt) o[nt][v] *= alpha;
      p0[v] = e0; p1[v] = e1;
    }

    // P (D-layout) -> per-wave LDS -> A-layout fragment (in-order DS, no barrier)
    unsigned short* pw = Pst[wave];
    #pragma unroll
    for (int v = 0; v < 8; ++v) {
      int m = sel * 8 + v;
      pw[m * 32 + lh]      = f2b(p0[v]);
      pw[m * 32 + 16 + lh] = f2b(p1[v]);
    }
    asm volatile("" ::: "memory");
    Frag ap;
    #pragma unroll
    for (int v = 0; v < 8; ++v) {
      int ka = sel * 16 + v * 2;
      ap.u[v] = *(const unsigned int*)&pw[lh * 32 + ka];
    }

    // O += P V  (B-frag pairs of consecutive keys are contiguous in Vt)
    #pragma unroll
    for (int nt = 0; nt < 8; ++nt) {
      Frag bv;
      #pragma unroll
      for (int v = 0; v < 8; ++v)
        bv.u[v] = *(const unsigned int*)&Vt[(nt * 16 + lh) * 32 + sel * 16 + v * 2];
      o[nt] = __builtin_amdgcn_wmma_f32_16x16x32_bf16(false, ap.v, false, bv.v, (short)0, o[nt], false, false);
    }
  }

  // normalize + write f32 output [B, T, H]
  float invl[8];
  #pragma unroll
  for (int v = 0; v < 8; ++v) invl[v] = 1.0f / lsum[v];
  #pragma unroll
  for (int nt = 0; nt < 8; ++nt)
    #pragma unroll
    for (int v = 0; v < 8; ++v) {
      size_t row = (size_t)b * T + q0 + sel * 8 + v;
      out[row * H + nt * 16 + lh] = o[nt][v] * invl[v];
    }
}

// ---------------------------------------------------------------------------
extern "C" void kernel_launch(void* const* d_in, const int* in_sizes, int n_in,
                              void* d_out, int out_size, void* d_ws, size_t ws_size,
                              hipStream_t stream) {
  const float* x  = (const float*)d_in[0];
  const float* Wk = (const float*)d_in[1];
  const float* Wq = (const float*)d_in[2];
  const float* Wv = (const float*)d_in[3];
  float* out = (float*)d_out;

  // workspace: bf16 K|Q|V (3*65536*128*2 = 48 MB), then bf16 W (0.75 MB)
  unsigned short* qkv = (unsigned short*)d_ws;
  unsigned int*   wbf = (unsigned int*)(qkv + (size_t)3 * BT * H);

  // Pass 0: convert weights to bf16 once (3*H*C/2 uint pairs)
  cvt_w_kernel<<<dim3(3 * H * C / 2 / 256), 256, 0, stream>>>(Wk, Wq, Wv, wbf);

  // Pass 1: 4096 M-tiles, 4 per block (8 waves, 2 waves per M-tile)
  qkv_proj_kernel<<<dim3(BT / 16 / 4), 256, 0, stream>>>(x, wbf, qkv);

  // Pass 2: one block per (64 query rows, batch)
  flash_attn_kernel<<<dim3(T / 64, Bc), 128, 0, stream>>>(qkv, out);
}
